// Attention_79207786873625
// MI455X (gfx1250) — matrix operand order repaired and checked by reading, not compiled, not run on hardware
//
#include <hip/hip_runtime.h>
#include <hip/hip_bf16.h>
#include <cstdint>

typedef __attribute__((ext_vector_type(16))) __bf16 v16bf;
typedef __attribute__((ext_vector_type(8)))  float  v8f;
typedef uint32_t u32x4 __attribute__((ext_vector_type(4)));
typedef int      i32x4 __attribute__((ext_vector_type(4)));
typedef int      i32x8 __attribute__((ext_vector_type(8)));

union FragBF {
  uint32_t u[8];
  uint4    q[2];
  v16bf    v;
};

__device__ __forceinline__ float bf2f(uint16_t s) {
  union { uint32_t u; float f; } o; o.u = ((uint32_t)s) << 16; return o.f;
}
__device__ __forceinline__ uint16_t f2bf(float f) {
  union { float f; uint32_t u; } v; v.f = f;
  uint32_t r = v.u + 0x7FFFu + ((v.u >> 16) & 1u);
  return (uint16_t)(r >> 16);
}
// A-matrix (16-bit, 16x32) per-reg K-pair start: V0:K0,1|16,17 V1:K2,3|18,19 ...
__device__ __forceinline__ int a_base(int r, int hi) {
  int b = 2 * r;
  return b + (hi ? 16 : 0);
}

// ---- Tensor Data Mover: 2-D tile load Global->LDS (D# per ISA 08 §8.3/8.4) ----
#if __has_builtin(__builtin_amdgcn_tensor_load_to_lds)
#define USE_TDM 1
__device__ __forceinline__ void tdm_load_2d(uint32_t ldsOff, const void* gaddr,
                                            uint32_t td0, uint32_t td1,
                                            uint32_t tile0, uint32_t tile1,
                                            uint64_t st0 /* elements */) {
  uint64_t ga = (uint64_t)(uintptr_t)gaddr;
  u32x4 g0;
  g0[0] = 1u;                                             // count=1, user mode
  g0[1] = ldsOff;                                         // lds_addr
  g0[2] = (uint32_t)ga;                                   // global_addr[31:0]
  g0[3] = (uint32_t)((ga >> 32) & 0x1FFFFFFu) | (2u << 30); // addr[56:32] | type=2
  i32x8 g1;
  g1[0] = (int)(1u << 16);                                // wg_mask=0, data_size=2B
  g1[1] = (int)((td0 & 0xFFFFu) << 16);                   // tensor_dim0[15:0]
  g1[2] = (int)(((td0 >> 16) & 0xFFFFu) | ((td1 & 0xFFFFu) << 16));
  g1[3] = (int)(((td1 >> 16) & 0xFFFFu) | ((tile0 & 0xFFFFu) << 16));
  g1[4] = (int)(tile1 & 0xFFFFu);                         // tile_dim1, tile_dim2=0
  g1[5] = (int)(uint32_t)(st0 & 0xFFFFFFFFu);             // dim0_stride[31:0]
  g1[6] = (int)(uint32_t)((st0 >> 32) & 0xFFFFu);         // dim0_stride[47:32]
  g1[7] = 0;
  i32x4 z4 = {0, 0, 0, 0};
#if defined(__clang_major__) && (__clang_major__ >= 23)
  i32x8 z8 = {0, 0, 0, 0, 0, 0, 0, 0};
  __builtin_amdgcn_tensor_load_to_lds(g0, g1, z4, z4, z8, 0);
#else
  __builtin_amdgcn_tensor_load_to_lds(g0, g1, z4, z4, 0);
#endif
}
__device__ __forceinline__ void tdm_wait0() {
#if __has_builtin(__builtin_amdgcn_s_wait_tensorcnt)
  __builtin_amdgcn_s_wait_tensorcnt(0);
#else
  asm volatile("s_wait_tensorcnt 0x0" ::: "memory");
#endif
}
#else
#define USE_TDM 0
#endif

// ---------------- kernel 0: weights f32 -> bf16 ----------------
__global__ void k_wconv(const float* __restrict__ wt, const float* __restrict__ wp,
                        const float* __restrict__ wg, const float* __restrict__ wo,
                        uint16_t* owt, uint16_t* owp, uint16_t* owg, uint16_t* owo) {
  int i = blockIdx.x * blockDim.x + threadIdx.x;
  if (i < 8192)        owt[i]          = f2bf(wt[i]);
  else if (i < 16384)  owp[i - 8192]   = f2bf(wp[i - 8192]);
  else if (i < 49152)  owg[i - 16384]  = f2bf(wg[i - 16384]);
  else if (i < 81920)  owo[i - 49152]  = f2bf(wo[i - 49152]);
}

// ---------------- kernel 1: 1x1-conv projections via WMMA ----------------
// x:[16,256,64,64] f32 -> Q:[b][n][32], PhiF:[b][n][32], GF:[b][n][128] (bf16)
__global__ void __launch_bounds__(256) k_proj(
    const float* __restrict__ x,
    const uint16_t* __restrict__ wth, const uint16_t* __restrict__ wph,
    const uint16_t* __restrict__ wgw,
    uint16_t* __restrict__ Q, uint16_t* __restrict__ PhiF, uint16_t* __restrict__ GF)
{
  __shared__ __align__(16) uint16_t xlds[64 * 256];  // [n][c] bf16, 32KB
  const int b   = blockIdx.x >> 6;
  const int n0  = (blockIdx.x & 63) * 64;
  const int tid = threadIdx.x;
  const int lane = tid & 31, wave = tid >> 5;
  const int col = lane & 15, hi = lane >> 4;

  { // coalesced f32 load of x tile, transpose-convert into LDS [n][c]
    const float4* xp = (const float4*)(x + ((size_t)(b * 256 + tid)) * 4096 + n0);
    #pragma unroll
    for (int j = 0; j < 16; ++j) {
      float4 f = xp[j];
      int n = j * 4;
      xlds[(n + 0) * 256 + tid] = f2bf(f.x);
      xlds[(n + 1) * 256 + tid] = f2bf(f.y);
      xlds[(n + 2) * 256 + tid] = f2bf(f.z);
      xlds[(n + 3) * 256 + tid] = f2bf(f.w);
    }
  }
  __syncthreads();

  // 48 output tiles (theta 8, phi 8, g 32); 8 waves x 6 tiles
  for (int i = 0; i < 6; ++i) {
    int t = wave * 6 + i;
    const uint16_t* wsrc; uint16_t* dst; int oc; int tt;
    if (t < 8)       { tt = t;      wsrc = wth; dst = Q;    oc = 32;  }
    else if (t < 16) { tt = t - 8;  wsrc = wph; dst = PhiF; oc = 32;  }
    else             { tt = t - 16; wsrc = wgw; dst = GF;   oc = 128; }
    const int ot = tt >> 2, nt = tt & 3;

    v8f acc = {0.f,0.f,0.f,0.f,0.f,0.f,0.f,0.f};
    #pragma unroll
    for (int k = 0; k < 8; ++k) {
      FragBF fa, fb;
      const uint16_t* wr = wsrc + (size_t)(ot * 16 + col) * 256 + k * 32;
      #pragma unroll
      for (int r = 0; r < 8; ++r)
        fa.u[r] = *(const uint32_t*)(wr + a_base(r, hi));
      const uint4* bp = (const uint4*)&xlds[(nt * 16 + col) * 256 + k * 32 + (hi ? 16 : 0)];
      fb.q[0] = bp[0]; fb.q[1] = bp[1];
      acc = __builtin_amdgcn_wmma_f32_16x16x32_bf16(false, fa.v, false, fb.v,
                                                    (short)0, acc, false, false);
    }
    uint16_t pk[8];
    #pragma unroll
    for (int v = 0; v < 8; ++v) pk[v] = f2bf(acc[v]);
    const int n = n0 + nt * 16 + col;
    uint16_t* out = dst + ((size_t)b * 4096 + n) * oc + ot * 16 + (hi ? 8 : 0);
    *(uint4*)out = *(const uint4*)pk;   // D cols -> contiguous channel run
  }
}

// ---------------- kernel 2: 2x2 maxpool -> K:[b][m][32], Vcm:[b][c][m] ----------------
__global__ void k_pool(const uint16_t* __restrict__ PhiF, const uint16_t* __restrict__ GF,
                       uint16_t* __restrict__ Kp, uint16_t* __restrict__ Vcm)
{
  const int NK = 16 * 1024 * 32;
  const int NV = 16 * 128 * 1024;
  int i = blockIdx.x * blockDim.x + threadIdx.x;
  if (i < NK) {
    int c = i & 31, m = (i >> 5) & 1023, b = i >> 15;
    int h2 = m >> 5, w2 = m & 31;
    int n00 = (h2 * 2) * 64 + w2 * 2;
    const uint16_t* src = PhiF + (size_t)b * 4096 * 32 + c;
    float v0 = bf2f(src[(size_t)(n00)      * 32]);
    float v1 = bf2f(src[(size_t)(n00 + 1)  * 32]);
    float v2 = bf2f(src[(size_t)(n00 + 64) * 32]);
    float v3 = bf2f(src[(size_t)(n00 + 65) * 32]);
    Kp[i] = f2bf(fmaxf(fmaxf(v0, v1), fmaxf(v2, v3)));
  } else if (i < NK + NV) {
    int j = i - NK;
    int m = j & 1023, c = (j >> 10) & 127, b = j >> 17;
    int h2 = m >> 5, w2 = m & 31;
    int n00 = (h2 * 2) * 64 + w2 * 2;
    const uint16_t* src = GF + (size_t)b * 4096 * 128 + c;
    float v0 = bf2f(src[(size_t)(n00)      * 128]);
    float v1 = bf2f(src[(size_t)(n00 + 1)  * 128]);
    float v2 = bf2f(src[(size_t)(n00 + 64) * 128]);
    float v3 = bf2f(src[(size_t)(n00 + 65) * 128]);
    Vcm[((size_t)b * 128 + c) * 1024 + m] = f2bf(fmaxf(fmaxf(v0, v1), fmaxf(v2, v3)));
  }
}

// ---------------- kernel 3: flash attention, WMMA + TDM staging ----------------
// Q:[b][4096][32], Kp:[b][1024][32], Vcm:[b][128][1024] -> Ows:[b][4096][128] (bf16)
__global__ void __launch_bounds__(256) k_attn(
    const uint16_t* __restrict__ Q, const uint16_t* __restrict__ Kp,
    const uint16_t* __restrict__ Vcm, uint16_t* __restrict__ Ows)
{
  __shared__ __align__(16) uint16_t Klds[128 * 32];     //  8KB [m][c]
  __shared__ __align__(16) uint16_t Vlds[128 * 128];    // 32KB [c][m]
  __shared__ __align__(16) uint16_t Plds[8][16 * 32];   //  8KB per-wave P [n][m]

  const int b    = blockIdx.x >> 5;
  const int nblk = blockIdx.x & 31;
  const int tid = threadIdx.x, lane = tid & 31, wave = tid >> 5;
  const int col = lane & 15, hi = lane >> 4;
  const int n0 = nblk * 128 + wave * 16;

  // Q A-fragment: full 32-channel contraction = exactly one WMMA K-step
  FragBF qa;
  {
    const uint16_t* qr = Q + ((size_t)b * 4096 + n0 + col) * 32;
    #pragma unroll
    for (int r = 0; r < 8; ++r) qa.u[r] = *(const uint32_t*)(qr + a_base(r, hi));
  }

  v8f oacc[8];
  #pragma unroll
  for (int ct = 0; ct < 8; ++ct)
    #pragma unroll
    for (int v = 0; v < 8; ++v) oacc[ct][v] = 0.f;
  float rm[8], l[8];
  #pragma unroll
  for (int v = 0; v < 8; ++v) { rm[v] = -3.0e38f; l[v] = 0.f; }

  for (int mc = 0; mc < 8; ++mc) {
    const int m0 = mc * 128;
    __syncthreads();
#if USE_TDM
    if (wave == 0) {
      // K chunk: 2-D tile [tile1=128 m][tile0=32 c] of tensor [1024][32]
      tdm_load_2d((uint32_t)(uintptr_t)&Klds[0],
                  Kp + ((size_t)b * 1024 + m0) * 32,
                  32u, 1024u, 32u, 128u, 32u);
      // V chunk: 2-D tile [tile1=128 c][tile0=128 m] of tensor [128][1024]
      tdm_load_2d((uint32_t)(uintptr_t)&Vlds[0],
                  Vcm + (size_t)b * 128 * 1024 + m0,
                  1024u, 128u, 128u, 128u, 1024u);
      tdm_wait0();
    }
#else
    { // K chunk [128 m][32 c]: 2 threads per row, 16B each
      int row = tid >> 1, half = tid & 1;
      const uint4* src = (const uint4*)(Kp + ((size_t)b * 1024 + m0 + row) * 32 + half * 16);
      uint4* dst = (uint4*)&Klds[row * 32 + half * 16];
      dst[0] = src[0];
    }
    { // V chunk [128 c][128 m]: 2 threads per row, 128B each
      int c = tid >> 1, half = tid & 1;
      const uint4* src = (const uint4*)(Vcm + ((size_t)b * 128 + c) * 1024 + m0 + half * 64);
      uint4* dst = (uint4*)&Vlds[c * 128 + half * 64];
      #pragma unroll
      for (int j = 0; j < 8; ++j) dst[j] = src[j];
    }
#endif
    __syncthreads();
    if (mc + 1 < 8)
      __builtin_prefetch(Kp + ((size_t)b * 1024 + m0 + 128) * 32, 0, 0);

    for (int ms = 0; ms < 4; ++ms) {     // 32-m step
      v8f s0, s1;
      {
        FragBF kb;
        const uint4* kp = (const uint4*)&Klds[((ms * 2 + 0) * 16 + col) * 32 + (hi ? 16 : 0)];
        kb.q[0] = kp[0]; kb.q[1] = kp[1];
        v8f z = {0.f,0.f,0.f,0.f,0.f,0.f,0.f,0.f};
        s0 = __builtin_amdgcn_wmma_f32_16x16x32_bf16(false, qa.v, false, kb.v,
                                                     (short)0, z, false, false);
      }
      {
        FragBF kb;
        const uint4* kp = (const uint4*)&Klds[((ms * 2 + 1) * 16 + col) * 32 + (hi ? 16 : 0)];
        kb.q[0] = kp[0]; kb.q[1] = kp[1];
        v8f z = {0.f,0.f,0.f,0.f,0.f,0.f,0.f,0.f};
        s1 = __builtin_amdgcn_wmma_f32_16x16x32_bf16(false, qa.v, false, kb.v,
                                                     (short)0, z, false, false);
      }
      // online softmax across the 32-m block; row = v+(hi?8:0) in D layout;
      // 16-lane xor reduction stays inside each half-wave (cols of the tile)
      #pragma unroll
      for (int v = 0; v < 8; ++v) {
        float mx = fmaxf(s0[v], s1[v]);
        mx = fmaxf(mx, __shfl_xor(mx, 1, 32));
        mx = fmaxf(mx, __shfl_xor(mx, 2, 32));
        mx = fmaxf(mx, __shfl_xor(mx, 4, 32));
        mx = fmaxf(mx, __shfl_xor(mx, 8, 32));
        float nm = fmaxf(rm[v], mx);
        float corr = __expf(rm[v] - nm);
        float p0 = __expf(s0[v] - nm);
        float p1 = __expf(s1[v] - nm);
        float ps = p0 + p1;
        ps += __shfl_xor(ps, 1, 32);
        ps += __shfl_xor(ps, 2, 32);
        ps += __shfl_xor(ps, 4, 32);
        ps += __shfl_xor(ps, 8, 32);
        l[v] = l[v] * corr + ps;
        rm[v] = nm;
        #pragma unroll
        for (int ct = 0; ct < 8; ++ct) oacc[ct][v] *= corr;
        const int prow = (v + (hi ? 8 : 0)) * 32;
        Plds[wave][prow + col]      = f2bf(p0);
        Plds[wave][prow + 16 + col] = f2bf(p1);
      }
      // re-layout P (D -> A) through per-wave LDS; DS ops are in-order per wave
      FragBF pa;
      {
        const uint16_t* pr = &Plds[wave][col * 32];
        #pragma unroll
        for (int r = 0; r < 8; ++r) pa.u[r] = *(const uint32_t*)(pr + a_base(r, hi));
      }
      #pragma unroll
      for (int ct = 0; ct < 8; ++ct) {
        FragBF vb;
        const uint4* vp = (const uint4*)&Vlds[(ct * 16 + col) * 128 + ms * 32 + (hi ? 16 : 0)];
        vb.q[0] = vp[0]; vb.q[1] = vp[1];
        oacc[ct] = __builtin_amdgcn_wmma_f32_16x16x32_bf16(false, pa.v, false, vb.v,
                                                           (short)0, oacc[ct], false, false);
      }
    }
  }

  float inv[8];
  #pragma unroll
  for (int v = 0; v < 8; ++v) inv[v] = 1.0f / l[v];
  #pragma unroll
  for (int ct = 0; ct < 8; ++ct) {
    const int c = ct * 16 + col;
    #pragma unroll
    for (int v = 0; v < 8; ++v) {
      const int n = n0 + v + (hi ? 8 : 0);
      Ows[((size_t)b * 4096 + n) * 128 + c] = f2bf(oacc[ct][v] * inv[v]);
    }
  }
}

// ---------------- kernel 4: w_o projection + gamma*o + x ----------------
__global__ void __launch_bounds__(256) k_out(
    const uint16_t* __restrict__ wob, const uint16_t* __restrict__ Ows,
    const float* __restrict__ x, const float* __restrict__ gamma,
    float* __restrict__ out)
{
  const int lane = threadIdx.x & 31, wave = threadIdx.x >> 5;
  const int col = lane & 15, hi = lane >> 4;
  const int g = blockIdx.x * 8 + wave;
  const int b = g >> 12;             // 16 co-tiles * 256 n-tiles per batch
  const int rem = g & 4095;
  const int cot = rem >> 8, nt = rem & 255;
  const float gm = gamma[0];

  v8f acc = {0.f,0.f,0.f,0.f,0.f,0.f,0.f,0.f};
  #pragma unroll
  for (int k = 0; k < 4; ++k) {
    FragBF fa, fb;
    const uint16_t* wr = wob + (size_t)(cot * 16 + col) * 128 + k * 32;
    #pragma unroll
    for (int r = 0; r < 8; ++r)
      fa.u[r] = *(const uint32_t*)(wr + a_base(r, hi));
    const uint4* bp = (const uint4*)(Ows + ((size_t)b * 4096 + nt * 16 + col) * 128
                                     + k * 32 + (hi ? 16 : 0));
    fb.q[0] = bp[0]; fb.q[1] = bp[1];
    acc = __builtin_amdgcn_wmma_f32_16x16x32_bf16(false, fa.v, false, fb.v,
                                                  (short)0, acc, false, false);
  }
  const int n = nt * 16 + col;
  #pragma unroll
  for (int v = 0; v < 8; ++v) {
    const int co = cot * 16 + v + (hi ? 8 : 0);
    const size_t idx = ((size_t)(b * 256 + co)) * 4096 + n;
    out[idx] = gm * acc[v] + x[idx];
  }
}

extern "C" void kernel_launch(void* const* d_in, const int* in_sizes, int n_in,
                              void* d_out, int out_size, void* d_ws, size_t ws_size,
                              hipStream_t stream) {
  const float* x     = (const float*)d_in[0];
  const float* wt    = (const float*)d_in[1];
  const float* wp    = (const float*)d_in[2];
  const float* wg    = (const float*)d_in[3];
  const float* wo    = (const float*)d_in[4];
  const float* gamma = (const float*)d_in[5];
  float* out = (float*)d_out;

  char* ws = (char*)d_ws;
  size_t off = 0;
  auto alloc = [&](size_t bytes) -> void* {
    void* p = ws + off;
    off = (off + bytes + 255) & ~(size_t)255;
    return p;
  };
  uint16_t* Q    = (uint16_t*)alloc((size_t)16 * 4096 * 32  * 2);   // 4  MiB
  uint16_t* PhiF = (uint16_t*)alloc((size_t)16 * 4096 * 32  * 2);   // 4  MiB
  uint16_t* GF   = (uint16_t*)alloc((size_t)16 * 4096 * 128 * 2);   // 16 MiB
  uint16_t* Kp   = (uint16_t*)alloc((size_t)16 * 1024 * 32  * 2);   // 1  MiB
  uint16_t* Vcm  = (uint16_t*)alloc((size_t)16 * 128 * 1024 * 2);   // 4  MiB
  uint16_t* Ows  = (uint16_t*)alloc((size_t)16 * 4096 * 128 * 2);   // 16 MiB
  uint16_t* wtb  = (uint16_t*)alloc(8192  * 2);
  uint16_t* wpb  = (uint16_t*)alloc(8192  * 2);
  uint16_t* wgb  = (uint16_t*)alloc(32768 * 2);
  uint16_t* wob  = (uint16_t*)alloc(32768 * 2);

  k_wconv<<<320, 256, 0, stream>>>(wt, wp, wg, wo, wtb, wpb, wgb, wob);
  k_proj<<<1024, 256, 0, stream>>>(x, wtb, wpb, wgb, Q, PhiF, GF);
  k_pool<<<(16 * 1024 * 32 + 16 * 128 * 1024 + 255) / 256, 256, 0, stream>>>(PhiF, GF, Kp, Vcm);
  k_attn<<<512, 256, 0, stream>>>(Q, Kp, Vcm, Ows);
  k_out<<<8192, 256, 0, stream>>>(wob, Ows, x, gamma, out);
}
